// HetGAT_4148938408770
// MI455X (gfx1250) — matrix operand order repaired and checked by reading, not compiled, not run on hardware
//
#include <hip/hip_runtime.h>
#include <hip/hip_bf16.h>
#include <math.h>

typedef __attribute__((ext_vector_type(16))) _Float16 v16h;
typedef __attribute__((ext_vector_type(8)))  _Float16 v8h;
typedef __attribute__((ext_vector_type(4)))  _Float16 v4h;
typedef __attribute__((ext_vector_type(8)))  float    v8f;
typedef __attribute__((ext_vector_type(4)))  float    v4f;

#define ACT_NONE 0
#define ACT_ELU  2

static const int NN = 20000;   // nodes per type
static const int EE = 400000;  // edges per relation
static const int HH = 10;      // heads

template<int ACT> __device__ __forceinline__ float act_apply(float v) {
    if constexpr (ACT == ACT_ELU) return v > 0.f ? v : (expf(v) - 1.f);
    return v;
}

// ---------------------------------------------------------------------------
// WMMA GEMM: C[M,N] = act(A[M,K] @ B[K,N] + bias[N])
// f32 in/out, f16 compute via v_wmma_f32_16x16x32_f16, f32 accumulate.
// Block: 256 threads = 8 waves; tile 128(M) x 64(N), BK=32.
// Waves stacked in M; each wave computes a 16x64 slab = 4 accumulators,
// 4 WMMAs per K-step from 1 A fragment + 4 B fragments.
// Requires K % 32 == 0 (true for all shapes here: 128,160,320,512).
// Staging is branch-free: OOB rows/cols are clamped; the corresponding
// outputs are never stored, so clamped garbage is harmless.
// ---------------------------------------------------------------------------
template<int ACT, bool HAS_BIAS>
__global__ __launch_bounds__(256)
void gemm_wmma_kernel(const float* __restrict__ A, const float* __restrict__ B,
                      const float* __restrict__ bias, float* __restrict__ C,
                      int M, int K, int N)
{
    __shared__ _Float16 As[128 * 32];   // [m][k]
    __shared__ _Float16 Bs[64 * 32];    // [n][k] (transposed on load)

    const int bm   = blockIdx.y * 128;
    const int bn   = blockIdx.x * 64;
    const int tid  = threadIdx.x;
    const int lane = tid & 31;
    const int wave = tid >> 5;          // 0..7, M direction

    v8f acc[4] = {{}, {}, {}, {}};
    const bool nvec = ((N & 3) == 0);

    for (int k0 = 0; k0 < K; k0 += 32) {
        // ---- stage A: 128x32 f32 -> f16, 4 x float4 per thread ----
#pragma unroll
        for (int i = 0; i < 4; ++i) {
            int lin = (tid + i * 256) << 2;        // 0..4092, multiple of 4
            int r = lin >> 5, c = lin & 31;        // r:0..127, c:0..28 step4
            int gr = bm + r; gr = gr < M ? gr : M - 1;
            v4f va = *(const v4f*)(A + (size_t)gr * K + (k0 + c));
            v4h ha;
            ha[0] = (_Float16)va[0]; ha[1] = (_Float16)va[1];
            ha[2] = (_Float16)va[2]; ha[3] = (_Float16)va[3];
            *(v4h*)&As[r * 32 + c] = ha;
        }
        // ---- stage B: 32x64 f32 -> Bs[n][k] f16, 2 x float4 per thread ----
#pragma unroll
        for (int i = 0; i < 2; ++i) {
            int lin = (tid + i * 256) << 2;        // 0..2044
            int kk = lin >> 6, nn = lin & 63;      // kk:0..31, nn:0..60 step4
            const float* Brow = B + (size_t)(k0 + kk) * N;
            v4f vb;
            if (nvec && (bn + nn + 3) < N) {
                vb = *(const v4f*)(Brow + bn + nn);
            } else {
#pragma unroll
                for (int j = 0; j < 4; ++j) {
                    int cc = bn + nn + j; cc = cc < N ? cc : N - 1;
                    vb[j] = Brow[cc];
                }
            }
#pragma unroll
            for (int j = 0; j < 4; ++j)
                Bs[(nn + j) * 32 + kk] = (_Float16)vb[j];
        }
        __syncthreads();

        // A fragment (16x32 f16): lane L holds row (L&15); halves 0..7 are
        // K = kb..kb+7, halves 8..15 are K = kb+16..kb+23, kb = (L>>4)*8.
        const int ar  = wave * 16 + (lane & 15);
        const int akb = (lane >> 4) << 3;
        v8h alo = *(const v8h*)&As[ar * 32 + akb];
        v8h ahi = *(const v8h*)&As[ar * 32 + akb + 16];
        v16h af = __builtin_shufflevector(alo, ahi,
            0,1,2,3,4,5,6,7,8,9,10,11,12,13,14,15);

        // B fragments (32x16 f16): lane L holds col (L&15); halves 0..7 are
        // K = kb..kb+7, halves 8..15 are K = kb+16..kb+23, kb = (L>>4)*8.
        const int bkb = (lane >> 4) << 3;
#pragma unroll
        for (int t = 0; t < 4; ++t) {
            int nc = t * 16 + (lane & 15);
            v8h bl = *(const v8h*)&Bs[nc * 32 + bkb];
            v8h bh = *(const v8h*)&Bs[nc * 32 + bkb + 16];
            v16h bf = __builtin_shufflevector(bl, bh,
                0,1,2,3,4,5,6,7,8,9,10,11,12,13,14,15);
            acc[t] = __builtin_amdgcn_wmma_f32_16x16x32_f16(
                false, af, false, bf, (short)0, acc[t], false, false);
        }
        __syncthreads();
    }

    // Epilogue. C/D layout: VGPR r -> M = r + (lane>>4)*8, N = lane&15.
    const int row0 = bm + wave * 16 + ((lane >> 4) << 3);
#pragma unroll
    for (int t = 0; t < 4; ++t) {
        int col = bn + t * 16 + (lane & 15);
        if (col < N) {
            float bv = HAS_BIAS ? bias[col] : 0.f;
#pragma unroll
            for (int r = 0; r < 8; ++r) {
                int row = row0 + r;
                if (row < M)
                    C[(size_t)row * N + col] = act_apply<ACT>(acc[t][r] + bv);
            }
        }
    }
}

// ---------------------------------------------------------------------------
// Small helper kernels
// ---------------------------------------------------------------------------
__global__ void k_fill(float* __restrict__ p, float v, size_t n) {
    size_t i = (size_t)blockIdx.x * blockDim.x + threadIdx.x;
    if (i < n) p[i] = v;
}

// W[hD, hD] block-diagonal from rel[H, D, D]
__global__ void k_blockdiag(const float* __restrict__ rel, float* __restrict__ W,
                            int H, int D) {
    int hd  = H * D;
    int idx = blockIdx.x * blockDim.x + threadIdx.x;
    if (idx >= hd * hd) return;
    int row = idx / hd, col = idx % hd;
    int h  = row / D, d = row % D;
    int h2 = col / D, e = col % D;
    W[idx] = (h == h2) ? rel[((size_t)h * D + d) * D + e] : 0.f;
}

__global__ void k_gelu(float* __restrict__ x, size_t n) {   // tanh approx (jax default)
    size_t i = (size_t)blockIdx.x * blockDim.x + threadIdx.x;
    if (i < n) {
        float v = x[i];
        float c = 0.7978845608028654f * (v + 0.044715f * v * v * v);
        x[i] = 0.5f * v * (1.f + tanhf(c));
    }
}

__device__ __forceinline__ void atomicMaxF(float* addr, float val) {
    unsigned int* ua = (unsigned int*)addr;
    unsigned int old = *ua;
    while (__uint_as_float(old) < val) {
        unsigned int assumed = old;
        old = atomicCAS(ua, assumed, __float_as_uint(val));
        if (old == assumed) break;
    }
}

// logit[e,h] = dot(q[dst,h,:], kr[src,h,:]) * scale * prel[h] + ea[e]*ew[h] + eb[h]
__global__ void k_edge_logit(const int* __restrict__ ei, const float* __restrict__ q,
                             const float* __restrict__ kr, const float* __restrict__ ea,
                             const float* __restrict__ prel, const float* __restrict__ ew,
                             const float* __restrict__ eb, float* __restrict__ logit,
                             int E, int H, int D, float scale)
{
    int idx = blockIdx.x * blockDim.x + threadIdx.x;
    if (idx >= E * H) return;
    int e = idx / H, h = idx % H;
    int s = ei[e], d = ei[E + e];
    const float* qp = q  + ((size_t)d * H + h) * D;   // 16-float aligned (D=16/32)
    const float* kp = kr + ((size_t)s * H + h) * D;
    float acc = 0.f;
    for (int i = 0; i < D; i += 4) {
        v4f qv = *(const v4f*)(qp + i);
        v4f kv = *(const v4f*)(kp + i);
        acc += qv[0]*kv[0] + qv[1]*kv[1] + qv[2]*kv[2] + qv[3]*kv[3];
    }
    logit[idx] = acc * scale * prel[h] + ea[e] * ew[h] + eb[h];
}

__global__ void k_seg_max(const int* __restrict__ ei, const float* __restrict__ logit,
                          float* __restrict__ m, int E, int H) {
    int idx = blockIdx.x * blockDim.x + threadIdx.x;
    if (idx >= E * H) return;
    int e = idx / H, h = idx % H;
    int d = ei[E + e];
    atomicMaxF(&m[(size_t)d * H + h], logit[idx]);
}

__global__ void k_exp_sum(const int* __restrict__ ei, float* __restrict__ logit,
                          const float* __restrict__ m, float* __restrict__ s,
                          int E, int H) {
    int idx = blockIdx.x * blockDim.x + threadIdx.x;
    if (idx >= E * H) return;
    int e = idx / H, h = idx % H;
    int d = ei[E + e];
    float x = expf(logit[idx] - m[(size_t)d * H + h]);
    logit[idx] = x;
    atomicAdd(&s[(size_t)d * H + h], x);
}

__global__ void k_agg(const int* __restrict__ ei, const float* __restrict__ logit,
                      const float* __restrict__ ssum, const float* __restrict__ vr,
                      float* __restrict__ agg, int E, int H, int D) {
    int idx = blockIdx.x * blockDim.x + threadIdx.x;
    if (idx >= E * H) return;
    int e = idx / H, h = idx % H;
    int sN = ei[e], d = ei[E + e];
    float a = logit[idx] / (ssum[(size_t)d * H + h] + 1e-16f);
    const float* vp = vr  + ((size_t)sN * H + h) * D;
    float*       op = agg + ((size_t)d  * H + h) * D;
    for (int i = 0; i < D; i += 4) {
        v4f vv = *(const v4f*)(vp + i);
        atomicAdd(&op[i + 0], vv[0] * a);
        atomicAdd(&op[i + 1], vv[1] * a);
        atomicAdd(&op[i + 2], vv[2] * a);
        atomicAdd(&op[i + 3], vv[3] * a);
    }
}

// BatchNorm (biased var, eps 1e-5), in-place; one block per column.
__global__ __launch_bounds__(256)
void k_bn(float* __restrict__ x, const float* __restrict__ g,
          const float* __restrict__ b, int M, int Ncols) {
    int col = blockIdx.x, tid = threadIdx.x;
    __shared__ float ss[256], sq[256];
    float s0 = 0.f, s1 = 0.f;
    for (int r = tid; r < M; r += 256) {
        float v = x[(size_t)r * Ncols + col];
        s0 += v; s1 += v * v;
    }
    ss[tid] = s0; sq[tid] = s1; __syncthreads();
    for (int o = 128; o > 0; o >>= 1) {
        if (tid < o) { ss[tid] += ss[tid + o]; sq[tid] += sq[tid + o]; }
        __syncthreads();
    }
    float mean = ss[0] / M;
    float var  = sq[0] / M - mean * mean;
    float inv  = rsqrtf(var + 1e-5f);
    float gg = g[col], bb = b[col];
    for (int r = tid; r < M; r += 256) {
        size_t ix = (size_t)r * Ncols + col;
        x[ix] = (x[ix] - mean) * inv * gg + bb;
    }
}

// out[row,:] = x[row,:] / sqrt(sum(x^2) + 1e-12); Ncols == blockDim.x (128)
__global__ __launch_bounds__(128)
void k_rownorm(const float* __restrict__ x, float* __restrict__ out, int Ncols) {
    int row = blockIdx.x, tid = threadIdx.x;
    __shared__ float sh[128];
    float v = x[(size_t)row * Ncols + tid];
    sh[tid] = v * v; __syncthreads();
    for (int o = 64; o > 0; o >>= 1) {
        if (tid < o) sh[tid] += sh[tid + o];
        __syncthreads();
    }
    out[(size_t)row * Ncols + tid] = v * rsqrtf(sh[0] + 1e-12f);
}

// row-wise softmax; one block per row
__global__ __launch_bounds__(256)
void k_softmax(const float* __restrict__ x, float* __restrict__ out, int Ncols) {
    int row = blockIdx.x, tid = threadIdx.x;
    __shared__ float sh[256];
    float m = -3.0e38f;
    for (int c = tid; c < Ncols; c += 256) m = fmaxf(m, x[(size_t)row * Ncols + c]);
    sh[tid] = m; __syncthreads();
    for (int o = 128; o > 0; o >>= 1) {
        if (tid < o) sh[tid] = fmaxf(sh[tid], sh[tid + o]);
        __syncthreads();
    }
    m = sh[0]; __syncthreads();
    float s = 0.f;
    for (int c = tid; c < Ncols; c += 256) s += expf(x[(size_t)row * Ncols + c] - m);
    sh[tid] = s; __syncthreads();
    for (int o = 128; o > 0; o >>= 1) {
        if (tid < o) sh[tid] += sh[tid + o];
        __syncthreads();
    }
    s = sh[0];
    for (int c = tid; c < Ncols; c += 256)
        out[(size_t)row * Ncols + c] = expf(x[(size_t)row * Ncols + c] - m) / s;
}

// ---------------------------------------------------------------------------
// Host-side orchestration
// ---------------------------------------------------------------------------
static inline int cdiv(int a, int b) { return (a + b - 1) / b; }

static void gemm(const float* A, const float* B, const float* bias, float* C,
                 int M, int K, int N, int act, hipStream_t st) {
    dim3 g(cdiv(N, 64), cdiv(M, 128)), b(256);
    if (bias) {
        if (act == ACT_ELU)
            gemm_wmma_kernel<ACT_ELU, true><<<g, b, 0, st>>>(A, B, bias, C, M, K, N);
        else
            gemm_wmma_kernel<ACT_NONE, true><<<g, b, 0, st>>>(A, B, bias, C, M, K, N);
    } else {
        gemm_wmma_kernel<ACT_NONE, false><<<g, b, 0, st>>>(A, B, nullptr, C, M, K, N);
    }
}

struct Scr {
    float *k_a, *q_a, *v_a, *k_u, *q_u, *v_u;
    float *kr, *vr, *agg_a, *agg_u;
    float *logits, *m, *s, *Wbd, *tmpO;
};

struct Hgt {
    const float *kw, *kb, *qw, *qb, *vw, *vb;
    const float *arel, *mrel, *prel, *ew, *eb, *alw, *alb;
};

static Hgt get_hgt(void* const* d_in, int base) {
    Hgt p;
    p.kw   = (const float*)d_in[base + 0];  p.kb   = (const float*)d_in[base + 1];
    p.qw   = (const float*)d_in[base + 2];  p.qb   = (const float*)d_in[base + 3];
    p.vw   = (const float*)d_in[base + 4];  p.vb   = (const float*)d_in[base + 5];
    p.arel = (const float*)d_in[base + 6];  p.mrel = (const float*)d_in[base + 7];
    p.prel = (const float*)d_in[base + 8];  p.ew   = (const float*)d_in[base + 9];
    p.eb   = (const float*)d_in[base + 10]; p.alw  = (const float*)d_in[base + 11];
    p.alb  = (const float*)d_in[base + 12];
    return p;
}

static void relation(const float* k_s, const float* v_s, const float* q_d,
                     const int* ei, const float* ea,
                     const float* arel_r, const float* mrel_r,
                     const float* prel_r, const float* ew_r, const float* eb_r,
                     float* agg, int D, Scr& S, hipStream_t st) {
    const int hd = HH * D;
    const int bdN = hd * hd;
    // relation transforms as block-diagonal WMMA GEMMs
    k_blockdiag<<<cdiv(bdN, 256), 256, 0, st>>>(arel_r, S.Wbd, HH, D);
    gemm(k_s, S.Wbd, nullptr, S.kr, NN, hd, hd, ACT_NONE, st);
    k_blockdiag<<<cdiv(bdN, 256), 256, 0, st>>>(mrel_r, S.Wbd, HH, D);
    gemm(v_s, S.Wbd, nullptr, S.vr, NN, hd, hd, ACT_NONE, st);

    size_t nh = (size_t)NN * HH;
    k_fill<<<cdiv((int)nh, 256), 256, 0, st>>>(S.m, -3.0e38f, nh);
    k_fill<<<cdiv((int)nh, 256), 256, 0, st>>>(S.s, 0.f, nh);
    size_t na = (size_t)NN * hd;
    k_fill<<<(int)((na + 255) / 256), 256, 0, st>>>(agg, 0.f, na);

    const int eh = EE * HH;
    float scale = 1.f / sqrtf((float)D);
    k_edge_logit<<<cdiv(eh, 256), 256, 0, st>>>(ei, q_d, S.kr, ea, prel_r, ew_r,
                                                eb_r, S.logits, EE, HH, D, scale);
    k_seg_max<<<cdiv(eh, 256), 256, 0, st>>>(ei, S.logits, S.m, EE, HH);
    k_exp_sum<<<cdiv(eh, 256), 256, 0, st>>>(ei, S.logits, S.m, S.s, EE, HH);
    k_agg<<<cdiv(eh, 256), 256, 0, st>>>(ei, S.logits, S.s, S.vr, agg, EE, HH, D);
}

static void hgt_layer(const float* xa, const float* xu, int fin, int D, const Hgt& p,
                      const int* ei_au, const float* ea_au,
                      const int* ei_ua, const float* ea_ua,
                      float* out_a, float* out_u, Scr& S, hipStream_t st) {
    const int hd = HH * D;
    const size_t woff = (size_t)fin * hd;
    gemm(xa, p.kw,        p.kb,      S.k_a, NN, fin, hd, ACT_NONE, st);
    gemm(xa, p.qw,        p.qb,      S.q_a, NN, fin, hd, ACT_NONE, st);
    gemm(xa, p.vw,        p.vb,      S.v_a, NN, fin, hd, ACT_NONE, st);
    gemm(xu, p.kw + woff, p.kb + hd, S.k_u, NN, fin, hd, ACT_NONE, st);
    gemm(xu, p.qw + woff, p.qb + hd, S.q_u, NN, fin, hd, ACT_NONE, st);
    gemm(xu, p.vw + woff, p.vb + hd, S.v_u, NN, fin, hd, ACT_NONE, st);

    const size_t roff = (size_t)HH * D * D;
    // r=0: ant -> user (k/v from ant, q from user, output agg_u)
    relation(S.k_a, S.v_a, S.q_u, ei_au, ea_au, p.arel, p.mrel,
             p.prel, p.ew, p.eb, S.agg_u, D, S, st);
    // r=1: user -> ant
    relation(S.k_u, S.v_u, S.q_a, ei_ua, ea_ua, p.arel + roff, p.mrel + roff,
             p.prel + HH, p.ew + HH, p.eb + HH, S.agg_a, D, S, st);

    size_t na = (size_t)NN * hd;
    k_gelu<<<(int)((na + 255) / 256), 256, 0, st>>>(S.agg_a, na);
    k_gelu<<<(int)((na + 255) / 256), 256, 0, st>>>(S.agg_u, na);
    // fuse the top-level ELU into the per-type output linear
    gemm(S.agg_a, p.alw,                    p.alb,      out_a, NN, hd, hd, ACT_ELU, st);
    gemm(S.agg_u, p.alw + (size_t)hd * hd,  p.alb + hd, out_u, NN, hd, hd, ACT_ELU, st);
}

extern "C" void kernel_launch(void* const* d_in, const int* in_sizes, int n_in,
                              void* d_out, int out_size, void* d_ws, size_t ws_size,
                              hipStream_t stream) {
    const float* x_ant  = (const float*)d_in[0];
    const float* x_user = (const float*)d_in[1];
    const float* ea_au  = (const float*)d_in[2];
    const float* ea_ua  = (const float*)d_in[3];
    Hgt l1 = get_hgt(d_in, 4);
    Hgt l2 = get_hgt(d_in, 17);
    const float* lin1_w  = (const float*)d_in[30];
    const float* lin1_b  = (const float*)d_in[31];
    const float* bn_g    = (const float*)d_in[32];
    const float* bn_b    = (const float*)d_in[33];
    const float* lin2_wa = (const float*)d_in[34];
    const float* lin2_ba = (const float*)d_in[35];
    const float* lin2_wu = (const float*)d_in[36];
    const float* lin2_bu = (const float*)d_in[37];
    const int*   ei_au   = (const int*)d_in[38];
    const int*   ei_ua   = (const int*)d_in[39];
    float* out = (float*)d_out;

    // ---- workspace carve-out (floats) ----
    float* W = (float*)d_ws;
    size_t off = 0;
    auto alloc = [&](size_t n) { float* p = W + off; off += n; return p; };
    float* ping_a = alloc((size_t)NN * 512);
    float* ping_u = alloc((size_t)NN * 512);
    float* pong_a = alloc((size_t)NN * 512);
    float* pong_u = alloc((size_t)NN * 512);
    Scr S;
    S.k_a = alloc((size_t)NN * 320); S.q_a = alloc((size_t)NN * 320);
    S.v_a = alloc((size_t)NN * 320); S.k_u = alloc((size_t)NN * 320);
    S.q_u = alloc((size_t)NN * 320); S.v_u = alloc((size_t)NN * 320);
    S.kr    = alloc((size_t)NN * 320); S.vr    = alloc((size_t)NN * 320);
    S.agg_a = alloc((size_t)NN * 320); S.agg_u = alloc((size_t)NN * 320);
    S.logits = alloc((size_t)EE * HH);
    S.m = alloc((size_t)NN * HH); S.s = alloc((size_t)NN * HH);
    S.Wbd  = alloc((size_t)320 * 320);
    S.tmpO = alloc((size_t)NN * 129);
    (void)ws_size; (void)in_sizes; (void)n_in; (void)out_size;

    // ---- layer 1 (F_IN=128, D=16) -> ping, ELU fused ----
    hgt_layer(x_ant, x_user, 128, 16, l1, ei_au, ea_au, ei_ua, ea_ua,
              ping_a, ping_u, S, stream);
    // ---- layer 2 (fin=160, D=32) -> pong, ELU fused ----
    hgt_layer(ping_a, ping_u, 160, 32, l2, ei_au, ea_au, ei_ua, ea_ua,
              pong_a, pong_u, S, stream);

    // ---- lin1 + ELU -> ping (reuse) ----
    gemm(pong_a, lin1_w,                     lin1_b,       ping_a, NN, 320, 512, ACT_ELU, stream);
    gemm(pong_u, lin1_w + (size_t)320 * 512, lin1_b + 512, ping_u, NN, 320, 512, ACT_ELU, stream);

    // ---- batchnorm (in-place) ----
    k_bn<<<512, 256, 0, stream>>>(ping_a, bn_g,       bn_b,       NN, 512);
    k_bn<<<512, 256, 0, stream>>>(ping_u, bn_g + 512, bn_b + 512, NN, 512);

    // ---- lin2 + heads ----
    gemm(ping_a, lin2_wa, lin2_ba, S.tmpO, NN, 512, 128, ACT_NONE, stream);
    k_rownorm<<<NN, 128, 0, stream>>>(S.tmpO, out, 128);
    gemm(ping_u, lin2_wu, lin2_bu, S.tmpO, NN, 512, 129, ACT_NONE, stream);
    k_softmax<<<NN, 256, 0, stream>>>(S.tmpO, out + (size_t)NN * 128, 129);
}
